// Block_7679401526028
// MI455X (gfx1250) — matrix operand order repaired and checked by reading, not compiled, not run on hardware
//
#include <hip/hip_runtime.h>
#include <hip/hip_bf16.h>
#include <cstdint>

typedef _Float16 f16;
typedef __attribute__((ext_vector_type(16))) _Float16 v16h;
typedef __attribute__((ext_vector_type(8)))  _Float16 v8h;
typedef __attribute__((ext_vector_type(8)))  float    v8f;

#define DEVINL __device__ __forceinline__

DEVINL v8h ld8(const f16* p) { return *reinterpret_cast<const v8h*>(p); }
DEVINL v16h cat(v8h a, v8h b) {
  return __builtin_shufflevector(a, b, 0,1,2,3,4,5,6,7,8,9,10,11,12,13,14,15);
}
DEVINL v8f wmma_f16(v16h a, v16h b, v8f c) {
  // D = A(16x32 f16) x B(32x16 f16) + C(16x16 f32)
  return __builtin_amdgcn_wmma_f32_16x16x32_f16(false, a, false, b, (short)0, c,
                                                false, false);
}

// ---------------------------------------------------------------------------
// Weight convert + transpose: Wt[n*K + k] = (f16) W[k*N + n]
// ---------------------------------------------------------------------------
__global__ __launch_bounds__(256) void conv_tr_kernel(const float* __restrict__ W,
                                                      f16* __restrict__ Wt,
                                                      int K, int N) {
  int idx = blockIdx.x * 256 + threadIdx.x;
  if (idx >= K * N) return;
  int k = idx / N, n = idx % N;
  Wt[(size_t)n * K + k] = (f16)W[(size_t)k * N + n];
}

// ---------------------------------------------------------------------------
// LayerNorm over C=768, one wave (32 lanes) per row, f16 output
// ---------------------------------------------------------------------------
__global__ __launch_bounds__(256) void ln_kernel(const float* __restrict__ x,
                                                 const float* __restrict__ w,
                                                 f16* __restrict__ out) {
  const int lane = threadIdx.x & 31;
  const int wave = threadIdx.x >> 5;
  const int row  = blockIdx.x * 8 + wave;
  const float* xr = x + (size_t)row * 768;
  float s = 0.f, s2 = 0.f;
  for (int i = lane; i < 768; i += 32) {
    float v = xr[i];
    s += v; s2 += v * v;
  }
  #pragma unroll
  for (int off = 16; off >= 1; off >>= 1) {
    s  += __shfl_xor(s,  off, 32);
    s2 += __shfl_xor(s2, off, 32);
  }
  const float mu   = s * (1.f / 768.f);
  const float var  = s2 * (1.f / 768.f) - mu * mu;
  const float rstd = rsqrtf(var + 1e-5f);
  f16* orow = out + (size_t)row * 768;
  for (int i = lane; i < 768; i += 32)
    orow[i] = (f16)((xr[i] - mu) * rstd * w[i]);
}

// ---------------------------------------------------------------------------
// Row sum-of-squares for q and k: q2[i] = sum_d q[i,d]^2   (rows of 64 f16)
// ---------------------------------------------------------------------------
__global__ __launch_bounds__(256) void rowsq_kernel(const f16* __restrict__ q,
                                                    const f16* __restrict__ k,
                                                    float* __restrict__ q2,
                                                    float* __restrict__ k2,
                                                    int rows) {
  int i = blockIdx.x * 256 + threadIdx.x;
  if (i >= rows) return;
  const f16* pq = q + (size_t)i * 64;
  const f16* pk = k + (size_t)i * 64;
  float sq = 0.f, sk = 0.f;
  #pragma unroll 8
  for (int d = 0; d < 64; ++d) {
    float a = (float)pq[d]; sq += a * a;
    float b = (float)pk[d]; sk += b * b;
  }
  q2[i] = sq;
  k2[i] = sk;
}

// ---------------------------------------------------------------------------
// Generic WMMA GEMM: out[M x N] = A[M x K] (f16, row-major) * Wt[N x K]^T
// 8 waves/block; each wave a 32x64 tile (2 A-frags x 4 B-frags -> 8 WMMA/chunk,
// every B-fragment reused twice). K-chunks of 32.
// ---------------------------------------------------------------------------
enum { EP_QKV = 0, EP_GELU = 1, EP_RESID = 2 };

template <int EP>
__global__ __launch_bounds__(256) void gemm_wmma_kernel(
    const f16* __restrict__ A, const f16* __restrict__ Bt, int N, int K,
    const float* __restrict__ res, float* __restrict__ outf,
    f16* __restrict__ outh, f16* __restrict__ qb, f16* __restrict__ kb,
    f16* __restrict__ vtb) {
  const int lane = threadIdx.x & 31;
  const int wave = threadIdx.x >> 5;
  const int nl   = lane & 15;   // tile-local N (or M for A-frags)
  const int hf   = lane >> 4;   // half-wave selector
  const int row0 = blockIdx.y * 256 + wave * 32;
  const int col0 = blockIdx.x * 64;

  v8f acc[2][4] = {};

  const f16* Arow0 = A + (size_t)(row0 + nl) * K;
  const f16* Arow1 = A + (size_t)(row0 + 16 + nl) * K;
  const f16* B0    = Bt + (size_t)(col0 + nl) * K;

  for (int kc = 0; kc < K; kc += 32) {
    // A-frags: lane halves hold K = 8*hf..+8 and 16+8*hf..+8
    v16h af0 = cat(ld8(Arow0 + kc + 8 * hf), ld8(Arow0 + kc + 16 + 8 * hf));
    v16h af1 = cat(ld8(Arow1 + kc + 8 * hf), ld8(Arow1 + kc + 16 + 8 * hf));
    __builtin_prefetch(Arow0 + kc + 128, 0, 1);  // speculative stream-ahead
    #pragma unroll
    for (int nt = 0; nt < 4; ++nt) {
      // B-frag: lane = column n, K = 8*hf..+8 and 16+8*hf..+8 (Wt[n][k])
      const f16* Bp = B0 + (size_t)(nt * 16) * K + kc + 8 * hf;
      v16h bf = cat(ld8(Bp), ld8(Bp + 16));
      acc[0][nt] = wmma_f16(af0, bf, acc[0][nt]);
      acc[1][nt] = wmma_f16(af1, bf, acc[1][nt]);
    }
  }

  // Epilogue. C/D layout: elem r -> m = r + 8*hf, n = nl
  #pragma unroll
  for (int mt = 0; mt < 2; ++mt) {
    #pragma unroll
    for (int nt = 0; nt < 4; ++nt) {
      #pragma unroll
      for (int r = 0; r < 8; ++r) {
        const int gm = row0 + mt * 16 + r + 8 * hf;
        const int gn = col0 + nt * 16 + nl;
        float v = acc[mt][nt][r];
        if (EP == EP_QKV) {
          const int b = gm >> 11, tt = gm & 2047;
          if (gn < 768) {
            const int head = gn >> 6, d = gn & 63;
            qb[(((size_t)b * 12 + head) * 2048 + tt) * 64 + d] = (f16)v;
          } else if (gn < 1536) {
            const int c = gn - 768, head = c >> 6, d = c & 63;
            kb[(((size_t)b * 12 + head) * 2048 + tt) * 64 + d] = (f16)v;
          } else {  // v stored transposed: vt[b,h,d,t]
            const int c = gn - 1536, head = c >> 6, d = c & 63;
            vtb[(((size_t)b * 12 + head) * 64 + d) * 2048 + tt] = (f16)v;
          }
        } else if (EP == EP_GELU) {
          outh[(size_t)gm * N + gn] =
              (f16)(0.5f * v * (1.f + erff(v * 0.70710678118654752f)));
        } else {  // EP_RESID
          outf[(size_t)gm * N + gn] = res[(size_t)gm * N + gn] + v;
        }
      }
    }
  }
}

// ---------------------------------------------------------------------------
// Attention: scores = exp(-1/(2*sqrt(D)) * (q2 + k2 - 2 q.k)), causal, no norm
// grid = (T/256, B*H); 8 waves, each owns 32 q-rows (2 row-tiles) so every
// K-fragment and V-fragment feeds two WMMAs. Per-wave LDS S re-layout.
// ---------------------------------------------------------------------------
__global__ __launch_bounds__(256) void attn_kernel(
    const f16* __restrict__ q, const f16* __restrict__ k,
    const f16* __restrict__ vt, const float* __restrict__ q2,
    const float* __restrict__ k2, f16* __restrict__ y) {
  __shared__ f16 smem[8 * 32 * 40];  // 8 waves * 32 rows * pitch 40 f16 (80B)

  const int lane = threadIdx.x & 31;
  const int wave = threadIdx.x >> 5;
  const int nl   = lane & 15;
  const int hf   = lane >> 4;
  const int bh   = blockIdx.y;                 // b*12 + head
  const int q0   = blockIdx.x * 256 + wave * 32;

  const f16*   qBH  = q  + (size_t)bh * 2048 * 64;
  const f16*   kBH  = k  + (size_t)bh * 2048 * 64;
  const f16*   vBH  = vt + (size_t)bh * 64 * 2048;
  const float* q2BH = q2 + (size_t)bh * 2048;
  const float* k2BH = k2 + (size_t)bh * 2048;
  f16* sm = smem + wave * 32 * 40;

  // Q A-fragments: 2 row-tiles x 2 d-chunks, resident for the whole kernel
  v16h qf[2][2];
  float q2v[2][8];
  #pragma unroll
  for (int i = 0; i < 2; ++i) {
    const f16* Qrow = qBH + (size_t)(q0 + 16 * i + nl) * 64;
    qf[i][0] = cat(ld8(Qrow + 8 * hf),      ld8(Qrow + 16 + 8 * hf));
    qf[i][1] = cat(ld8(Qrow + 32 + 8 * hf), ld8(Qrow + 48 + 8 * hf));
    #pragma unroll
    for (int r = 0; r < 8; ++r) q2v[i][r] = q2BH[q0 + 16 * i + r + 8 * hf];
  }

  v8f Y[2][4] = {};
  const float scale = -0.0625f;  // -1/(2*sqrt(64))

  for (int kc = 0; kc < q0 + 32; kc += 32) {
    // --- S tile: 32 q-rows x 32 keys, built as two 16-key subtiles ---
    #pragma unroll
    for (int j = 0; j < 2; ++j) {
      const int t0 = kc + 16 * j;
      const f16* Krow = kBH + (size_t)(t0 + nl) * 64;  // B-frag: col = key
      v16h kf0 = cat(ld8(Krow + 8 * hf),      ld8(Krow + 16 + 8 * hf));
      v16h kf1 = cat(ld8(Krow + 32 + 8 * hf), ld8(Krow + 48 + 8 * hf));
      const float k2v = k2BH[t0 + nl];
      #pragma unroll
      for (int i = 0; i < 2; ++i) {
        v8f sacc = {};
        sacc = wmma_f16(qf[i][0], kf0, sacc);
        sacc = wmma_f16(qf[i][1], kf1, sacc);
        #pragma unroll
        for (int r = 0; r < 8; ++r) {
          const int m  = r + 8 * hf;
          const int tq = q0 + 16 * i + m;
          const int tk = t0 + nl;
          float sc = (tk <= tq)
                         ? __expf(scale * (q2v[i][r] + k2v - 2.f * sacc[r]))
                         : 0.f;
          sm[(16 * i + m) * 40 + 16 * j + nl] = (f16)sc;  // C-layout -> row-major
        }
      }
    }
    // wave-private LDS: DS ops in-order; wait for stores before re-layout read
    asm volatile("s_wait_dscnt 0" ::: "memory");
    v16h sf[2];
    #pragma unroll
    for (int i = 0; i < 2; ++i)
      sf[i] = cat(ld8(sm + (16 * i + nl) * 40 + 8 * hf),
                  ld8(sm + (16 * i + nl) * 40 + 16 + 8 * hf));
    asm volatile("" ::: "memory");
    // --- Y += S(32x32) * V(32x64) via pre-transposed vt[d][t] ---
    #pragma unroll
    for (int nt = 0; nt < 4; ++nt) {
      const f16* Vp = vBH + (size_t)(nt * 16 + nl) * 2048 + kc + 8 * hf;
      v16h vf = cat(ld8(Vp), ld8(Vp + 16));
      Y[0][nt] = wmma_f16(sf[0], vf, Y[0][nt]);
      Y[1][nt] = wmma_f16(sf[1], vf, Y[1][nt]);
    }
  }

  // write y as [token][C] f16 (A operand of the projection GEMM)
  const int b    = bh / 12;
  const int head = bh % 12;
  #pragma unroll
  for (int i = 0; i < 2; ++i) {
    #pragma unroll
    for (int nt = 0; nt < 4; ++nt) {
      #pragma unroll
      for (int r = 0; r < 8; ++r) {
        const int t = q0 + 16 * i + r + 8 * hf;
        const int c = head * 64 + nt * 16 + nl;
        y[((size_t)b * 2048 + t) * 768 + c] = (f16)Y[i][nt][r];
      }
    }
  }
}

// ---------------------------------------------------------------------------
extern "C" void kernel_launch(void* const* d_in, const int* in_sizes, int n_in,
                              void* d_out, int out_size, void* d_ws,
                              size_t ws_size, hipStream_t stream) {
  const float* x       = (const float*)d_in[0];
  const float* w_ln1   = (const float*)d_in[1];
  const float* w_attn  = (const float*)d_in[2];
  const float* w_aproj = (const float*)d_in[3];
  const float* w_ln2   = (const float*)d_in[4];
  const float* w_fc    = (const float*)d_in[5];
  const float* w_mlp   = (const float*)d_in[6];
  float* out = (float*)d_out;

  constexpr int Bb = 2, T = 2048, C = 768, H = 12;
  constexpr int M = Bb * T, C3 = 3 * C, C4 = 4 * C;

  char* p = (char*)d_ws;
  auto alloc = [&](size_t bytes) -> void* {
    void* r = (void*)p;
    p += (bytes + 255) & ~(size_t)255;
    return r;
  };
  f16*   wt_attn  = (f16*)alloc((size_t)C3 * C * 2);
  f16*   wt_aproj = (f16*)alloc((size_t)C * C * 2);
  f16*   wt_fc    = (f16*)alloc((size_t)C4 * C * 2);
  f16*   wt_mlp   = (f16*)alloc((size_t)C * C4 * 2);
  f16*   h1  = (f16*)alloc((size_t)M * C * 2);
  f16*   qb  = (f16*)alloc((size_t)M * C * 2);
  f16*   kb  = (f16*)alloc((size_t)M * C * 2);
  f16*   vtb = (f16*)alloc((size_t)M * C * 2);
  float* q2  = (float*)alloc((size_t)Bb * H * T * 4);
  float* k2  = (float*)alloc((size_t)Bb * H * T * 4);
  f16*   yb  = (f16*)alloc((size_t)M * C * 2);
  float* x1  = (float*)alloc((size_t)M * C * 4);
  f16*   h2  = (f16*)alloc((size_t)M * C * 2);
  f16*   fca = (f16*)alloc((size_t)M * C4 * 2);

  // 1) f16-transposed weights
  conv_tr_kernel<<<dim3((C * C3 + 255) / 256), 256, 0, stream>>>(w_attn, wt_attn, C, C3);
  conv_tr_kernel<<<dim3((C * C  + 255) / 256), 256, 0, stream>>>(w_aproj, wt_aproj, C, C);
  conv_tr_kernel<<<dim3((C * C4 + 255) / 256), 256, 0, stream>>>(w_fc, wt_fc, C, C4);
  conv_tr_kernel<<<dim3((C4 * C + 255) / 256), 256, 0, stream>>>(w_mlp, wt_mlp, C4, C);

  // 2) LN1
  ln_kernel<<<dim3(M / 8), 256, 0, stream>>>(x, w_ln1, h1);

  // 3) qkv = h1 @ w_attn, scatter to q/k/v^T (f16)
  gemm_wmma_kernel<EP_QKV><<<dim3(C3 / 64, M / 256), 256, 0, stream>>>(
      h1, wt_attn, C3, C, nullptr, nullptr, nullptr, qb, kb, vtb);

  // 4) q2 / k2
  rowsq_kernel<<<dim3((Bb * H * T + 255) / 256), 256, 0, stream>>>(qb, kb, q2, k2,
                                                                   Bb * H * T);

  // 5) attention
  attn_kernel<<<dim3(T / 256, Bb * H), 256, 0, stream>>>(qb, kb, vtb, q2, k2, yb);

  // 6) x1 = x + y @ w_attn_proj
  gemm_wmma_kernel<EP_RESID><<<dim3(C / 64, M / 256), 256, 0, stream>>>(
      yb, wt_aproj, C, C, x, x1, nullptr, nullptr, nullptr, nullptr);

  // 7) LN2
  ln_kernel<<<dim3(M / 8), 256, 0, stream>>>(x1, w_ln2, h2);

  // 8) fca = gelu(h2 @ w_fc)  (f16)
  gemm_wmma_kernel<EP_GELU><<<dim3(C4 / 64, M / 256), 256, 0, stream>>>(
      h2, wt_fc, C4, C, nullptr, nullptr, fca, nullptr, nullptr, nullptr);

  // 9) out = x1 + fca @ w_mlp_proj
  gemm_wmma_kernel<EP_RESID><<<dim3(C / 64, M / 256), 256, 0, stream>>>(
      fca, wt_mlp, C, C4, x1, out, nullptr, nullptr, nullptr, nullptr);
}
